// ContrastiveLoss_27504970563831
// MI455X (gfx1250) — compile-verified, matrix operand order checked by reading, not run on hardware
//
#include <hip/hip_runtime.h>
#include <hip/hip_bf16.h>
#include <math.h>

#define NROWS 4096
#define DDIM  1024
#define TEMP_INV (1.0f / 0.07f)

typedef __bf16 v8bf  __attribute__((ext_vector_type(8)));
typedef __bf16 v16bf __attribute__((ext_vector_type(16)));
typedef float  v8f   __attribute__((ext_vector_type(8)));

#define CAT16(lo8, hi8) \
  __builtin_shufflevector((lo8), (hi8), 0,1,2,3,4,5,6,7,8,9,10,11,12,13,14,15)

__device__ __forceinline__ float rowmax16(float v) {
    v = fmaxf(v, __shfl_xor(v, 1, 32));
    v = fmaxf(v, __shfl_xor(v, 2, 32));
    v = fmaxf(v, __shfl_xor(v, 4, 32));
    v = fmaxf(v, __shfl_xor(v, 8, 32));
    return v;
}
__device__ __forceinline__ float rowsum16(float v) {
    v += __shfl_xor(v, 1, 32);
    v += __shfl_xor(v, 2, 32);
    v += __shfl_xor(v, 4, 32);
    v += __shfl_xor(v, 8, 32);
    return v;
}

// ---------------------------------------------------------------------------
// Pass 1: L2-normalize one row, emit bf16 hi/lo split of the normalized row.
// grid = NROWS blocks, 256 threads. Each thread handles 4 strided elements.
// ---------------------------------------------------------------------------
__global__ void __launch_bounds__(256)
prep_kernel(const float* __restrict__ z, __bf16* __restrict__ h,
            __bf16* __restrict__ l) {
    const int row = blockIdx.x;
    const int t   = threadIdx.x;
    const float* zr = z + (size_t)row * DDIM;

    float v[4];
    float ss = 0.0f;
#pragma unroll
    for (int j = 0; j < 4; ++j) {
        v[j] = zr[t + 256 * j];
        ss += v[j] * v[j];
    }
    // wave32 reduction
#pragma unroll
    for (int m = 16; m >= 1; m >>= 1) ss += __shfl_xor(ss, m, 32);

    __shared__ float wsum[8];
    __shared__ float invS;
    if ((t & 31) == 0) wsum[t >> 5] = ss;
    __syncthreads();
    if (t == 0) {
        float s = 0.0f;
        for (int w = 0; w < 8; ++w) s += wsum[w];
        invS = 1.0f / fmaxf(sqrtf(s), 1e-12f);
    }
    __syncthreads();
    const float inv = invS;

#pragma unroll
    for (int j = 0; j < 4; ++j) {
        float x   = v[j] * inv;
        __bf16 hi = (__bf16)x;
        float lo  = x - (float)hi;
        size_t idx = (size_t)row * DDIM + t + 256 * j;
        h[idx] = hi;
        l[idx] = (__bf16)lo;
    }
}

// ---------------------------------------------------------------------------
// Pass 2: fused GEMM (bf16x3 WMMA) + online row softmax + diagonal capture.
// One block = 16 rows of logits. 8 waves, each owns a 16x16 tile per n-iter
// (16 x 128 logits per iteration, 32 iterations over N=4096 columns).
// ---------------------------------------------------------------------------
__global__ void __launch_bounds__(256)
fused_kernel(const __bf16* __restrict__ h1, const __bf16* __restrict__ l1,
             const __bf16* __restrict__ h2, const __bf16* __restrict__ l2,
             float* __restrict__ partial) {
    __shared__ float tileMax[8][16];
    __shared__ float tileSum[8][16];
    __shared__ float runMaxS[16];
    __shared__ float runSumS[16];
    __shared__ float diagS[16];

    const int tid  = threadIdx.x;
    const int wave = tid >> 5;
    const int lane = tid & 31;
    const int half = lane >> 4;   // selects K groups {0-7,16-23} vs {8-15,24-31}
    const int mr   = lane & 15;   // A row / B column within the 16x16 tile
    const int m0   = blockIdx.x * 16;

    if (tid < 16) {
        runMaxS[tid] = -__builtin_inff();
        runSumS[tid] = 0.0f;
        diagS[tid]   = 0.0f;
    }
    __syncthreads();

    // A-fragment base pointers (same 16 rows for every wave; L0/L2 resident)
    const __bf16* arow_h = h1 + (size_t)(m0 + mr) * DDIM + 8 * half;
    const __bf16* arow_l = l1 + (size_t)(m0 + mr) * DDIM + 8 * half;

    for (int n0 = 0; n0 < NROWS; n0 += 128) {
        const int ncol = n0 + wave * 16 + mr;
        const __bf16* brow_h = h2 + (size_t)ncol * DDIM + 8 * half;
        const __bf16* brow_l = l2 + (size_t)ncol * DDIM + 8 * half;

        v8f acc = {};
#pragma unroll 4
        for (int k0 = 0; k0 < DDIM; k0 += 32) {
            v8bf ah0 = *(const v8bf*)(arow_h + k0);
            v8bf ah1 = *(const v8bf*)(arow_h + k0 + 16);
            v8bf al0 = *(const v8bf*)(arow_l + k0);
            v8bf al1 = *(const v8bf*)(arow_l + k0 + 16);
            v8bf bh0 = *(const v8bf*)(brow_h + k0);
            v8bf bh1 = *(const v8bf*)(brow_h + k0 + 16);
            v8bf bl0 = *(const v8bf*)(brow_l + k0);
            v8bf bl1 = *(const v8bf*)(brow_l + k0 + 16);

            v16bf aH = CAT16(ah0, ah1);
            v16bf aL = CAT16(al0, al1);
            v16bf bH = CAT16(bh0, bh1);
            v16bf bL = CAT16(bl0, bl1);

            // bf16x3: hi*hi + hi*lo + lo*hi  (~fp32 accuracy)
            acc = __builtin_amdgcn_wmma_f32_16x16x32_bf16(
                false, aH, false, bH, (short)0, acc, false, false);
            acc = __builtin_amdgcn_wmma_f32_16x16x32_bf16(
                false, aH, false, bL, (short)0, acc, false, false);
            acc = __builtin_amdgcn_wmma_f32_16x16x32_bf16(
                false, aL, false, bH, (short)0, acc, false, false);
        }

        // C layout: VGPR r holds row m = 8*half + r, column n = mr.
        float lg[8];
#pragma unroll
        for (int r = 0; r < 8; ++r) lg[r] = acc[r] * TEMP_INV;

        // capture diagonal logit (each global row hits exactly one column)
#pragma unroll
        for (int r = 0; r < 8; ++r) {
            const int mg = m0 + 8 * half + r;
            if (ncol == mg) diagS[8 * half + r] = lg[r];
        }

        // per-row tile max (reduce across the 16 lanes of this half)
#pragma unroll
        for (int r = 0; r < 8; ++r) {
            float mx = rowmax16(lg[r]);
            if (mr == r) tileMax[wave][8 * half + r] = mx;
        }
        __syncthreads();

        // combine tile maxima, rescale running sums
        if (tid < 16) {
            float im = tileMax[0][tid];
#pragma unroll
            for (int w = 1; w < 8; ++w) im = fmaxf(im, tileMax[w][tid]);
            float nm = fmaxf(runMaxS[tid], im);
            runSumS[tid] *= __expf(runMaxS[tid] - nm);
            runMaxS[tid] = nm;
        }
        __syncthreads();

        // per-row exp-sums against the new running max
#pragma unroll
        for (int r = 0; r < 8; ++r) {
            float e = __expf(lg[r] - runMaxS[8 * half + r]);
            float s = rowsum16(e);
            if (mr == r) tileSum[wave][8 * half + r] = s;
        }
        __syncthreads();

        if (tid < 16) {
            float s = 0.0f;
#pragma unroll
            for (int w = 0; w < 8; ++w) s += tileSum[w][tid];
            runSumS[tid] += s;
        }
        __syncthreads();
    }

    if (tid == 0) {
        float s = 0.0f;
        for (int m = 0; m < 16; ++m) {
            // -log_softmax diagonal: -(diag - max - log(sumexp))
            s += -(diagS[m] - runMaxS[m] - __logf(runSumS[m]));
        }
        partial[blockIdx.x] = s;
    }
}

// ---------------------------------------------------------------------------
// Pass 3: deterministic fixed-order reduction of 256 per-block partials.
// ---------------------------------------------------------------------------
__global__ void __launch_bounds__(256)
finalize_kernel(const float* __restrict__ partial, float* __restrict__ out) {
    __shared__ float s[256];
    const int t = threadIdx.x;
    s[t] = partial[t];
    __syncthreads();
    for (int off = 128; off > 0; off >>= 1) {
        if (t < off) s[t] += s[t + off];
        __syncthreads();
    }
    if (t == 0) out[0] = s[0] / (float)NROWS;
}

// ---------------------------------------------------------------------------
extern "C" void kernel_launch(void* const* d_in, const int* in_sizes, int n_in,
                              void* d_out, int out_size, void* d_ws,
                              size_t ws_size, hipStream_t stream) {
    (void)in_sizes; (void)n_in; (void)out_size; (void)ws_size;

    const float* z1 = (const float*)d_in[0];
    const float* z2 = (const float*)d_in[1];
    float* out = (float*)d_out;

    const size_t elems = (size_t)NROWS * DDIM;  // 4M elements
    __bf16* h1 = (__bf16*)d_ws;                 //  8 MB
    __bf16* l1 = h1 + elems;                    //  8 MB
    __bf16* h2 = l1 + elems;                    //  8 MB
    __bf16* l2 = h2 + elems;                    //  8 MB
    float* partial = (float*)(l2 + elems);      //  1 KB (256 floats)

    prep_kernel<<<NROWS, 256, 0, stream>>>(z1, h1, l1);
    prep_kernel<<<NROWS, 256, 0, stream>>>(z2, h2, l2);
    fused_kernel<<<NROWS / 16, 256, 0, stream>>>(h1, l1, h2, l2, partial);
    finalize_kernel<<<1, 256, 0, stream>>>(partial, out);
}
